// AugmentedNeuralODE_90752658964593
// MI455X (gfx1250) — matrix operand order repaired and checked by reading, not compiled, not run on hardware
//
#include <hip/hip_runtime.h>
#include <cmath>

#define LATENT   123
#define TOTAL    128
#define HID      512
#define BATCH    1024
#define TSTEPS   8
#define SUBSTEPS 6

typedef __attribute__((ext_vector_type(16))) _Float16 v16h;
typedef __attribute__((ext_vector_type(8)))  _Float16 v8h;
typedef __attribute__((ext_vector_type(8)))  float    v8f;
typedef __attribute__((ext_vector_type(4)))  float    v4f;

union F16Frag { v16h v; v8h h[2]; };

// LDS strides (in elements), padded to avoid 64-bank conflicts
#define YIN_S 136   // halves, for 16x128 f16 stage input
#define H_S   520   // halves, for 16x512 f16 hidden activations
#define K_S   132   // floats, for 16x128 f32 GEMM3 output staging

__device__ __forceinline__ float fast_tanh(float x) {
#if __has_builtin(__builtin_amdgcn_tanhf)
  return __builtin_amdgcn_tanhf(x);
#else
  return tanhf(x);
#endif
}

__device__ __forceinline__ v8f wmma_f16(const F16Frag& a, const F16Frag& b, v8f c) {
  return __builtin_amdgcn_wmma_f32_16x16x32_f16(
      /*neg_a=*/false, a.v, /*neg_b=*/false, b.v,
      /*c_mod=*/(short)0, c, /*reuse_a=*/false, /*reuse_b=*/false);
}

// One GEMM layer: wave computes TPW adjacent 16x16 output tiles concurrently
// (TPW independent wmma chains), with an explicit 2-deep double-buffered
// K-pipeline (unroll disabled) to bound register pressure below 256 VGPRs.
// A in LDS (f16, row-major, stride a_stride halves).
// Wt uniform global base; per-lane 32-bit offsets -> saddr+voffset loads.
template<int KDIM, int TPW, bool DOTANH>
__device__ __forceinline__ void gemm_multi(
    const _Float16* __restrict__ Wt,
    const float*    __restrict__ bias,
    const _Float16* Ain, int a_stride,
    _Float16* outH, float* outF, int o_stride,
    int wave, int lane)
{
  constexpr int KT = KDIM / 32;          // k-iterations (power of two: 4 or 16)
  const int hi = lane >> 4;              // 0 or 1 (half-wave)
  const int nl = lane & 15;              // column within tile / row of A
  const int kA = hi * 8;                 // A-fragment K base (16-bit A 16x32 layout)
  const int kB = hi * 8;                 // B-fragment K base (16-bit B 32x16 layout)
  const int mb = hi * 8;                 // C/D row base (32-bit C/D 16x16 layout)

  v8f acc[TPW];
  unsigned woff[TPW];                    // 32-bit element offsets into Wt
  #pragma unroll
  for (int i = 0; i < TPW; ++i) {
    const int n = (wave * TPW + i) * 16 + nl;
    const float bv = bias[n];
    #pragma unroll
    for (int r = 0; r < 8; ++r) acc[i][r] = bv;
    woff[i] = (unsigned)n * KDIM + kB;
  }
  const _Float16* arow = Ain + nl * a_stride + kA;

  F16Frag a[2];
  F16Frag b[2][TPW];

  auto load_frags = [&](int buf, int kt) {
    a[buf].h[0] = *(const v8h*)(arow + kt * 32);        // K = base .. base+7
    a[buf].h[1] = *(const v8h*)(arow + kt * 32 + 16);   // K = base+16 .. base+23
    #pragma unroll
    for (int i = 0; i < TPW; ++i) {
      const unsigned o = woff[i] + (unsigned)kt * 32;
      b[buf][i].h[0] = *(const v8h*)(Wt + o);           // K = base .. base+7
      b[buf][i].h[1] = *(const v8h*)(Wt + o + 16);      // K = base+16 .. base+23
    }
  };

  load_frags(0, 0);
  #pragma clang loop unroll(disable)
  for (int kt = 0; kt < KT; kt += 2) {
    load_frags(1, kt + 1);
    #pragma unroll
    for (int i = 0; i < TPW; ++i) acc[i] = wmma_f16(a[0], b[0][i], acc[i]);
    load_frags(0, (kt + 2) & (KT - 1));   // branchless wrap for last prefetch
    #pragma unroll
    for (int i = 0; i < TPW; ++i) acc[i] = wmma_f16(a[1], b[1][i], acc[i]);
  }

  #pragma unroll
  for (int i = 0; i < TPW; ++i) {
    const int nc = (wave * TPW + i) * 16 + nl;
    #pragma unroll
    for (int r = 0; r < 8; ++r) {
      const int m = mb + r;
      if (DOTANH) outH[m * o_stride + nc] = (_Float16)fast_tanh(acc[i][r]);
      else        outF[m * o_stride + nc] = acc[i][r];
    }
  }
}

// One MLP evaluation: kout = f(yin_vals). yin_vals/kout are per-thread
// 8-element slices (thread owns row=tid/16, cols (tid%16)*8 .. +7).
__device__ __forceinline__ void feval(
    _Float16* s_yin, _Float16* s_h1, _Float16* s_h2, float* s_k,
    const _Float16* W1t, const _Float16* W2t, const _Float16* W3t,
    const float* b1, const float* b2, const float* b3,
    int wave, int lane, int row, int c0,
    const float yin_vals[8], float kout[8])
{
  v8h hv;
  #pragma unroll
  for (int j = 0; j < 8; ++j) hv[j] = (_Float16)yin_vals[j];
  *(v8h*)(s_yin + row * YIN_S + c0) = hv;
  __syncthreads();
  gemm_multi<TOTAL, 4, true >(W1t, b1, s_yin, YIN_S, s_h1, nullptr, H_S, wave, lane);
  __syncthreads();
  gemm_multi<HID,   4, true >(W2t, b2, s_h1, H_S, s_h2, nullptr, H_S, wave, lane);
  __syncthreads();
  gemm_multi<HID,   1, false>(W3t, b3, s_h2, H_S, nullptr, s_k, K_S, wave, lane);
  __syncthreads();
  const v4f* kp = (const v4f*)(s_k + row * K_S + c0);
  v4f a0 = kp[0], a1 = kp[1];
  #pragma unroll
  for (int j = 0; j < 4; ++j) { kout[j] = a0[j]; kout[4 + j] = a1[j]; }
}

// Transpose + f16-convert the three weight matrices into workspace.
// ws layout (halves): W1t [512][128] @0, W2t [512][512] @65536, W3t [128][512] @327680
__global__ void node_convert_weights(const float* __restrict__ W1,
                                     const float* __restrict__ W2,
                                     const float* __restrict__ W3,
                                     _Float16* __restrict__ ws)
{
  const int i = blockIdx.x * blockDim.x + threadIdx.x;
  if (i < 65536) {                       // W1t[n*128+k] = W1[k*512+n]
    const int n = i >> 7, k = i & 127;
    ws[i] = (_Float16)W1[k * 512 + n];
  } else if (i < 65536 + 262144) {       // W2t[n*512+k] = W2[k*512+n]
    const int j = i - 65536;
    const int n = j >> 9, k = j & 511;
    ws[i] = (_Float16)W2[k * 512 + n];
  } else if (i < 393216) {               // W3t[n*512+k] = W3[k*128+n]
    const int j = i - 327680;
    const int n = j >> 9, k = j & 511;
    ws[i] = (_Float16)W3[k * 128 + n];
  }
}

__global__ __launch_bounds__(256) void node_integrate(
    const float* __restrict__ z0, const float* __restrict__ tg,
    const float* __restrict__ b1, const float* __restrict__ b2,
    const float* __restrict__ b3,
    const _Float16* __restrict__ W1t, const _Float16* __restrict__ W2t,
    const _Float16* __restrict__ W3t,
    float* __restrict__ out)
{
  __shared__ _Float16 s_yin[16 * YIN_S];
  __shared__ _Float16 s_h1 [16 * H_S];
  __shared__ _Float16 s_h2 [16 * H_S];
  __shared__ float    s_k  [16 * K_S];

  const int tid  = threadIdx.x;
  const int wave = tid >> 5;
  const int lane = tid & 31;
  const int row  = tid >> 4;            // 0..15 : batch row within tile
  const int c0   = (tid & 15) * 8;      // first of 8 owned state columns
  const size_t bi = (size_t)blockIdx.x * 16 + row;

  // init state: first 123 cols from z0, augmented 5 cols zero
  float y[8];
  #pragma unroll
  for (int j = 0; j < 8; ++j) {
    const int c = c0 + j;
    y[j] = (c < LATENT) ? z0[bi * LATENT + c] : 0.0f;
  }
  { // output at t0
    float* op = out + (bi * TSTEPS + 0) * LATENT;
    #pragma unroll
    for (int j = 0; j < 8; ++j) { const int c = c0 + j; if (c < LATENT) op[c] = y[j]; }
  }

  float k1[8], k2[8], k3[8], k4[8], k5[8], k6[8], yt[8];

  for (int ti = 0; ti < TSTEPS - 1; ++ti) {
    const float dt = (tg[ti + 1] - tg[ti]) * (1.0f / SUBSTEPS);
    for (int ss = 0; ss < SUBSTEPS; ++ss) {
      // stage 1
      feval(s_yin, s_h1, s_h2, s_k, W1t, W2t, W3t, b1, b2, b3, wave, lane, row, c0, y, k1);
      // stage 2
      #pragma unroll
      for (int j = 0; j < 8; ++j) yt[j] = y[j] + dt * (0.2f * k1[j]);
      feval(s_yin, s_h1, s_h2, s_k, W1t, W2t, W3t, b1, b2, b3, wave, lane, row, c0, yt, k2);
      // stage 3
      #pragma unroll
      for (int j = 0; j < 8; ++j) yt[j] = y[j] + dt * (0.075f * k1[j] + 0.225f * k2[j]);
      feval(s_yin, s_h1, s_h2, s_k, W1t, W2t, W3t, b1, b2, b3, wave, lane, row, c0, yt, k3);
      // stage 4
      #pragma unroll
      for (int j = 0; j < 8; ++j)
        yt[j] = y[j] + dt * (0.9777777777777777f * k1[j] - 3.7333333333333334f * k2[j]
                           + 3.5555555555555554f * k3[j]);
      feval(s_yin, s_h1, s_h2, s_k, W1t, W2t, W3t, b1, b2, b3, wave, lane, row, c0, yt, k4);
      // stage 5
      #pragma unroll
      for (int j = 0; j < 8; ++j)
        yt[j] = y[j] + dt * (2.9525986892242035f * k1[j] - 11.595793324188385f * k2[j]
                           + 9.822892851699436f  * k3[j] - 0.2908093278463649f * k4[j]);
      feval(s_yin, s_h1, s_h2, s_k, W1t, W2t, W3t, b1, b2, b3, wave, lane, row, c0, yt, k5);
      // stage 6
      #pragma unroll
      for (int j = 0; j < 8; ++j)
        yt[j] = y[j] + dt * (2.8462752525252526f * k1[j] - 10.757575757575758f * k2[j]
                           + 8.906422717743473f  * k3[j] + 0.2784090909090909f * k4[j]
                           - 0.2735313036020583f * k5[j]);
      feval(s_yin, s_h1, s_h2, s_k, W1t, W2t, W3t, b1, b2, b3, wave, lane, row, c0, yt, k6);
      // 5th-order update
      #pragma unroll
      for (int j = 0; j < 8; ++j)
        y[j] += dt * (0.09114583333333333f * k1[j] + 0.4492362982929021f * k3[j]
                    + 0.6510416666666666f  * k4[j] - 0.322376179245283f  * k5[j]
                    + 0.13095238095238096f * k6[j]);
    }
    { // output at t_{ti+1}
      float* op = out + (bi * TSTEPS + (size_t)(ti + 1)) * LATENT;
      #pragma unroll
      for (int j = 0; j < 8; ++j) { const int c = c0 + j; if (c < LATENT) op[c] = y[j]; }
    }
  }
}

extern "C" void kernel_launch(void* const* d_in, const int* in_sizes, int n_in,
                              void* d_out, int out_size, void* d_ws, size_t ws_size,
                              hipStream_t stream) {
  const float* z0 = (const float*)d_in[0];
  const float* tg = (const float*)d_in[1];
  const float* W1 = (const float*)d_in[2];
  const float* b1 = (const float*)d_in[3];
  const float* W2 = (const float*)d_in[4];
  const float* b2 = (const float*)d_in[5];
  const float* W3 = (const float*)d_in[6];
  const float* b3 = (const float*)d_in[7];
  _Float16* ws = (_Float16*)d_ws;

  node_convert_weights<<<1536, 256, 0, stream>>>(W1, W2, W3, ws);
  node_integrate<<<BATCH / 16, 256, 0, stream>>>(
      z0, tg, b1, b2, b3, ws, ws + 65536, ws + 327680, (float*)d_out);
}
